// SessionModel_62414464746143
// MI455X (gfx1250) — matrix operand order repaired and checked by reading, not compiled, not run on hardware
//
#include <hip/hip_runtime.h>

#define VOCAB 100000
#define HID   128
#define BATCH 1024

typedef __bf16 v16bf __attribute__((ext_vector_type(16)));
typedef __bf16 v8bf  __attribute__((ext_vector_type(8)));
typedef __bf16 v4bf  __attribute__((ext_vector_type(4)));
typedef float  v8f   __attribute__((ext_vector_type(8)));

union V16U { v16bf v; v8bf h[2]; };

static __device__ inline v8f wmma_bf16(v16bf a, v16bf b, v8f c) {
    return __builtin_amdgcn_wmma_f32_16x16x32_bf16(
        /*neg_a=*/false, a, /*neg_b=*/false, b,
        /*c_mod=*/(short)0, c, /*reuse_a=*/false, /*reuse_b=*/false);
}

// ---------------------------------------------------------------------------
// Kernel 1: GRU cell step (negligible: ~100 MFLOP). Also emits bf16 hi/lo
// planes of h_new for the split-precision WMMA GEMM (fast path only).
// ---------------------------------------------------------------------------
__global__ __launch_bounds__(128) void gru_cell_kernel(
    const int* __restrict__ input, const float* __restrict__ hidden,
    const float* __restrict__ Wih, const float* __restrict__ bih,
    const float* __restrict__ Whh, const float* __restrict__ bhh,
    float* __restrict__ hnew,
    __bf16* __restrict__ ahi, __bf16* __restrict__ alo)
{
    __shared__ float h[HID];
    const int t = threadIdx.x;
    const int b = blockIdx.x;
    h[t] = hidden[b * HID + t];
    __syncthreads();

    const int idx = input[b];
    float gi_r = Wih[(size_t)(t          ) * VOCAB + idx] + bih[t          ];
    float gi_z = Wih[(size_t)(t +     HID) * VOCAB + idx] + bih[t +     HID];
    float gi_n = Wih[(size_t)(t + 2 * HID) * VOCAB + idx] + bih[t + 2 * HID];

    float gr = bhh[t], gz = bhh[t + HID], gn = bhh[t + 2 * HID];
#pragma unroll 8
    for (int k = 0; k < HID; ++k) {
        const float hk = h[k];
        gr += hk * Whh[(t          ) * HID + k];
        gz += hk * Whh[(t +     HID) * HID + k];
        gn += hk * Whh[(t + 2 * HID) * HID + k];
    }
    const float r = 1.0f / (1.0f + __expf(-(gi_r + gr)));
    const float z = 1.0f / (1.0f + __expf(-(gi_z + gz)));
    const float n = tanhf(gi_n + r * gn);
    const float hn = (1.0f - z) * n + z * h[t];
    hnew[b * HID + t] = hn;
    if (ahi) {
        const __bf16 hb = (__bf16)hn;
        ahi[b * HID + t] = hb;
        alo[b * HID + t] = (__bf16)(hn - (float)hb);
    }
}

// ---------------------------------------------------------------------------
// Kernel 2 (fast path prep): split W_out f32 -> bf16 hi/lo planes, one pass.
// 51.2 MB read + 51.2 MB write ~= 4.4 us at HBM rate; runs once per call.
// ---------------------------------------------------------------------------
__global__ __launch_bounds__(256) void split_wout_kernel(
    const float* __restrict__ W, __bf16* __restrict__ hi, __bf16* __restrict__ lo)
{
    const size_t i = ((size_t)blockIdx.x * blockDim.x + threadIdx.x) * 4;
    const float4 f = *(const float4*)(W + i);
    const float  ff[4] = {f.x, f.y, f.z, f.w};
    v4bf h, l;
#pragma unroll
    for (int j = 0; j < 4; ++j) {
        const __bf16 hb = (__bf16)ff[j];
        h[j] = hb;
        l[j] = (__bf16)(ff[j] - (float)hb);
    }
    *(v4bf*)(hi + i) = h;
    *(v4bf*)(lo + i) = l;
}

// ---------------------------------------------------------------------------
// Kernel 3 (fast path): logits = ReLU(h_new @ W_out.T + b_out) from
// pre-split bf16 planes. One wave computes a 16x80 strip (5 16x16 tiles).
// Hot loop = b128 loads + v_wmma only (no conversions).
// ---------------------------------------------------------------------------
__global__ __launch_bounds__(256) void gru_logits_wmma_pre_kernel(
    const __bf16* __restrict__ Ahi, const __bf16* __restrict__ Alo,
    const __bf16* __restrict__ Bhi, const __bf16* __restrict__ Blo,
    const float* __restrict__ bout, float* __restrict__ logit)
{
    const int lane = threadIdx.x;            // wave32
    const int hl   = lane >> 4;
    const int l16  = lane & 15;

    const int mb = (blockIdx.y * 8 + threadIdx.y) * 16;  // 64 M-tiles
    const int nb = blockIdx.x * 80;                      // 1250 * 80 = 100000

    // A (16x32 bf16): lanes 0-15 -> K 0..15; lanes 16-31 -> K 16..31
    const int sA = hl ? 16 : 0;
    // B (32x16 bf16): lanes 0-15 -> K 0..15; lanes 16-31 -> K 16..31; N = lane&15
    const int sB = hl ? 16 : 0;

    const size_t rowA = (size_t)(mb + l16) * HID + sA;
    size_t rowB[5];
#pragma unroll
    for (int t = 0; t < 5; ++t)
        rowB[t] = (size_t)(nb + t * 16 + l16) * HID + sB;

    v8f acc[5] = {};

#pragma unroll
    for (int kc = 0; kc < HID; kc += 32) {
        V16U ah, al;
        ah.h[0] = *(const v8bf*)(Ahi + rowA + kc);
        ah.h[1] = *(const v8bf*)(Ahi + rowA + kc + 8);
        al.h[0] = *(const v8bf*)(Alo + rowA + kc);
        al.h[1] = *(const v8bf*)(Alo + rowA + kc + 8);
#pragma unroll
        for (int t = 0; t < 5; ++t) {
            V16U bh, bl;
            bh.h[0] = *(const v8bf*)(Bhi + rowB[t] + kc);
            bh.h[1] = *(const v8bf*)(Bhi + rowB[t] + kc + 8);
            bl.h[0] = *(const v8bf*)(Blo + rowB[t] + kc);
            bl.h[1] = *(const v8bf*)(Blo + rowB[t] + kc + 8);
            // 3-term split-precision: ah*bh + ah*bl + al*bh (~fp32 accuracy)
            acc[t] = wmma_bf16(ah.v, bh.v, acc[t]);
            acc[t] = wmma_bf16(ah.v, bl.v, acc[t]);
            acc[t] = wmma_bf16(al.v, bh.v, acc[t]);
        }
    }

    // Epilogue: bias + ReLU + store (C/D layout: VGPR r -> row r + 8*half)
#pragma unroll
    for (int t = 0; t < 5; ++t) {
        const int n = nb + t * 16 + l16;
        const float bo = bout[n];
#pragma unroll
        for (int r = 0; r < 8; ++r) {
            const int m = mb + r + hl * 8;
            float v = acc[t][r] + bo;
            logit[(size_t)m * VOCAB + n] = v > 0.0f ? v : 0.0f;
        }
    }
}

// ---------------------------------------------------------------------------
// Fallback (workspace too small): convert-in-loop variant from round 1.
// ---------------------------------------------------------------------------
static __device__ inline void split16(const float* __restrict__ f,
                                      v16bf& hi, v16bf& lo) {
#pragma unroll
    for (int i = 0; i < 16; ++i) {
        const __bf16 h = (__bf16)f[i];
        hi[i] = h;
        lo[i] = (__bf16)(f[i] - (float)h);
    }
}

__global__ __launch_bounds__(256) void gru_logits_wmma_kernel(
    const float* __restrict__ hnew, const float* __restrict__ Wout,
    const float* __restrict__ bout, float* __restrict__ logit)
{
    const int lane = threadIdx.x;
    const int hl   = lane >> 4;
    const int l16  = lane & 15;

    const int mb  = (blockIdx.y * 8 + threadIdx.y) * 16;
    const int nb0 = blockIdx.x * 32;
    const int nb1 = nb0 + 16;

    const int sA = hl ? 16 : 0;
    const int sB = hl ? 16 : 0;

    const size_t rowA  = (size_t)(mb + l16) * HID;
    const size_t rowB0 = (size_t)(nb0 + l16) * HID;
    const size_t rowB1 = (size_t)(nb1 + l16) * HID;

    v8f acc0 = {};
    v8f acc1 = {};

#pragma unroll
    for (int kc = 0; kc < HID; kc += 32) {
        const float4* pa0 = (const float4*)(hnew + rowA + kc + sA);
        const float4* pa1 = (const float4*)(hnew + rowA + kc + sA + 8);
        const float4 a0 = pa0[0], a1 = pa0[1], a2 = pa1[0], a3 = pa1[1];
        const float af[16] = {a0.x, a0.y, a0.z, a0.w, a1.x, a1.y, a1.z, a1.w,
                              a2.x, a2.y, a2.z, a2.w, a3.x, a3.y, a3.z, a3.w};
        v16bf ah, al;
        split16(af, ah, al);

        const float4* pb0 = (const float4*)(Wout + rowB0 + kc + sB);
        const float4 b0 = pb0[0], b1 = pb0[1], b2 = pb0[2], b3 = pb0[3];
        const float bf0[16] = {b0.x, b0.y, b0.z, b0.w, b1.x, b1.y, b1.z, b1.w,
                               b2.x, b2.y, b2.z, b2.w, b3.x, b3.y, b3.z, b3.w};
        v16bf b0h, b0l;
        split16(bf0, b0h, b0l);

        const float4* pb1 = (const float4*)(Wout + rowB1 + kc + sB);
        const float4 c0 = pb1[0], c1 = pb1[1], c2 = pb1[2], c3 = pb1[3];
        const float bf1[16] = {c0.x, c0.y, c0.z, c0.w, c1.x, c1.y, c1.z, c1.w,
                               c2.x, c2.y, c2.z, c2.w, c3.x, c3.y, c3.z, c3.w};
        v16bf b1h, b1l;
        split16(bf1, b1h, b1l);

        acc0 = wmma_bf16(ah, b0h, acc0);
        acc0 = wmma_bf16(ah, b0l, acc0);
        acc0 = wmma_bf16(al, b0h, acc0);

        acc1 = wmma_bf16(ah, b1h, acc1);
        acc1 = wmma_bf16(ah, b1l, acc1);
        acc1 = wmma_bf16(al, b1h, acc1);
    }

    const int n0 = nb0 + l16;
    const int n1 = nb1 + l16;
    const float bo0 = bout[n0];
    const float bo1 = bout[n1];
#pragma unroll
    for (int r = 0; r < 8; ++r) {
        const int m = mb + r + hl * 8;
        float v0 = acc0[r] + bo0;
        float v1 = acc1[r] + bo1;
        logit[(size_t)m * VOCAB + n0] = v0 > 0.0f ? v0 : 0.0f;
        logit[(size_t)m * VOCAB + n1] = v1 > 0.0f ? v1 : 0.0f;
    }
}

// ---------------------------------------------------------------------------
extern "C" void kernel_launch(void* const* d_in, const int* in_sizes, int n_in,
                              void* d_out, int out_size, void* d_ws, size_t ws_size,
                              hipStream_t stream)
{
    (void)in_sizes; (void)n_in; (void)out_size;

    const int*   input  = (const int*)  d_in[0];
    const float* hidden = (const float*)d_in[1];
    const float* Wih    = (const float*)d_in[2];
    const float* bih    = (const float*)d_in[3];
    const float* Whh    = (const float*)d_in[4];
    const float* bhh    = (const float*)d_in[5];
    const float* Wout   = (const float*)d_in[6];
    const float* bout   = (const float*)d_in[7];

    float* logit = (float*)d_out;                              // (1024, 100000)
    float* hnew  = (float*)d_out + (size_t)BATCH * VOCAB;      // (1, 1024, 128)

    const size_t nW = (size_t)VOCAB * HID;   // 12.8M elements
    const size_t nA = (size_t)BATCH * HID;
    const size_t need = (2 * nW + 2 * nA) * sizeof(__bf16);    // ~51.75 MB

    if (ws_size >= need) {
        // Fast path: pre-split bf16 hi/lo planes in workspace.
        __bf16* Bhi = (__bf16*)d_ws;
        __bf16* Blo = Bhi + nW;
        __bf16* Ahi = Blo + nW;
        __bf16* Alo = Ahi + nA;

        gru_cell_kernel<<<BATCH, HID, 0, stream>>>(
            input, hidden, Wih, bih, Whh, bhh, hnew, Ahi, Alo);

        split_wout_kernel<<<(unsigned)(nW / (4 * 256)), 256, 0, stream>>>(
            Wout, Bhi, Blo);

        dim3 grid(VOCAB / 80, BATCH / 16 / 8);   // (1250, 8)
        dim3 block(32, 8);                       // 8 waves / block, 8 M-tiles
        gru_logits_wmma_pre_kernel<<<grid, block, 0, stream>>>(
            Ahi, Alo, Bhi, Blo, bout, logit);
    } else {
        // Fallback: convert-in-loop WMMA GEMM.
        gru_cell_kernel<<<BATCH, HID, 0, stream>>>(
            input, hidden, Wih, bih, Whh, bhh, hnew, nullptr, nullptr);

        dim3 grid(VOCAB / 32, BATCH / 16 / 8);   // (3125, 8)
        dim3 block(32, 8);
        gru_logits_wmma_kernel<<<grid, block, 0, stream>>>(
            hnew, Wout, bout, logit);
    }
}
